// ConfigurableGAT_53463752900646
// MI455X (gfx1250) — matrix operand order repaired and checked by reading, not compiled, not run on hardware
//
#include <hip/hip_runtime.h>
#include <hip/hip_bf16.h>

typedef unsigned short u16;
typedef unsigned int   u32;

typedef __attribute__((ext_vector_type(16))) __bf16 v16bf;
typedef __attribute__((ext_vector_type(8)))  float  v8f;

#define GAT_NEG_SLOPE 0.2f

// ---------- helpers ----------
__device__ __forceinline__ u16 f2bf(float f) {           // f32 -> bf16, RNE
  u32 u = __float_as_uint(f);
  u32 r = u + 0x7FFFu + ((u >> 16) & 1u);
  return (u16)(r >> 16);
}
// order-preserving float <-> uint map for atomicMax on floats
__device__ __forceinline__ u32 fenc(float f) {
  u32 u = __float_as_uint(f);
  return (u & 0x80000000u) ? ~u : (u | 0x80000000u);
}
__device__ __forceinline__ float fdec(u32 u) {
  u32 v = (u & 0x80000000u) ? (u & 0x7FFFFFFFu) : ~u;
  return __uint_as_float(v);
}
__device__ __forceinline__ void edge_sd(const int* __restrict__ ei, int E, int e,
                                        int& s, int& d) {
  if (e < E) { s = ei[e]; d = ei[E + e]; }     // row 0 = src, row 1 = dst
  else       { s = e - E; d = e - E; }          // self-loops appended
}

// ---------- small utility kernels ----------
__global__ void k_f32_to_bf16(const float* __restrict__ s, u16* __restrict__ d, int n) {
  int i = blockIdx.x * blockDim.x + threadIdx.x;
  if (i < n) d[i] = f2bf(s[i]);
}
__global__ void k_zero_f32(float* __restrict__ p, int n) {
  int i = blockIdx.x * blockDim.x + threadIdx.x;
  if (i < n) p[i] = 0.0f;
}
__global__ void k_init_enc(u32* __restrict__ p, int n) {
  int i = blockIdx.x * blockDim.x + threadIdx.x;
  if (i < n) p[i] = fenc(-__builtin_inff());
}

// ---------- W pre-pack: f32 [K x NOUT] -> bf16 B-fragment layout ----------
// Wp[((tn*(K/32) + kblk)*32 + lane)*8 + slot] packs rows (k0, k0+1) of column n:
//   lm = lane&15, lh = lane>>4, k0 = kblk*32 + lh*16 + slot*2, n = tn*16 + lm
// so each lane's 8 u32 for one wmma are 32 contiguous bytes (2x b128 loads).
template<int K, int NOUT>
__global__ void k_pack_w(const float* __restrict__ W, u32* __restrict__ Wp) {
  constexpr int total = (NOUT / 16) * (K / 32) * 32 * 8;
  int i = blockIdx.x * blockDim.x + threadIdx.x;
  if (i >= total) return;
  int slot = i & 7;
  int lane = (i >> 3) & 31;
  int kblk = (i >> 8) % (K / 32);
  int tn   = (i >> 8) / (K / 32);
  int lm = lane & 15, lh = lane >> 4;
  int k0 = kblk * 32 + lh * 16 + slot * 2;
  int n  = tn * 16 + lm;
  u32 lo = f2bf(W[(size_t)k0 * NOUT + n]);
  u32 hi = f2bf(W[(size_t)(k0 + 1) * NOUT + n]);
  Wp[i] = lo | (hi << 16);
}

// ---------- bf16 WMMA GEMM:  Out[M x NOUT] = X[M x K] * W[K x NOUT] ----------
// One wave computes a 16 x (16*NT) strip: A fragment loaded once per K-step
// (two b128s), reused for NT back-to-back v_wmma_f32_16x16x32_bf16 issues.
// Fragment layouts used here (wave32; A/B k order is contiguous, not the ISA table order):
//   A 16x32 bf16: lane m=lane&15, lh=lane>>4; VGPR i: K = lh*16 + 2i
//   B 32x16 bf16: lane n=lane&15; VGPR i: K = lh*16 + 2i   (pre-packed above)
//   C/D f32:      VGPR r -> row m = lh*8 + r, col n = lane&15
template<int K, int NOUT>
__global__ void k_gemm_bf16(const u16* __restrict__ X, const u32* __restrict__ Wp,
                            float* __restrict__ Out, int M) {
  constexpr int tilesN  = NOUT / 16;
  constexpr int NT      = (tilesN < 4) ? tilesN : 4;   // n-tiles per wave
  constexpr int groupsN = tilesN / NT;
  const int tilesM = (M + 15) >> 4;
  const int wave = blockIdx.x * (blockDim.x >> 5) + (threadIdx.x >> 5);
  if (wave >= tilesM * groupsN) return;
  const int tm = wave / groupsN;
  const int tg = wave % groupsN;
  const int lane = threadIdx.x & 31;
  const int lm = lane & 15;
  const int lh = lane >> 4;

  int arow = tm * 16 + lm;
  if (arow >= M) arow = M - 1;                          // tail clamp (dead for 50000)
  const u16* __restrict__ xrow = X + (size_t)arow * K;

  v8f acc[NT] = {};
#pragma unroll
  for (int kb = 0; kb < K; kb += 32) {
    union { u32 u[8]; v16bf v; } A;
    *(uint4*)&A.u[0] = *(const uint4*)(xrow + kb + lh * 16);       // K=kb+lh*16 .. +7
    *(uint4*)&A.u[4] = *(const uint4*)(xrow + kb + lh * 16 + 8);   // K=kb+lh*16+8 ..
#pragma unroll
    for (int t = 0; t < NT; ++t) {
      const int tn = tg * NT + t;
      const u32* __restrict__ bp =
          Wp + ((size_t)(tn * (K / 32) + (kb >> 5)) * 32 + lane) * 8;
      union { u32 u[8]; v16bf v; } B;
      *(uint4*)&B.u[0] = *(const uint4*)(bp);
      *(uint4*)&B.u[4] = *(const uint4*)(bp + 4);
      acc[t] = __builtin_amdgcn_wmma_f32_16x16x32_bf16(false, A.v, false, B.v,
                                                       (short)0, acc[t], false, false);
    }
  }

  const int mbase = tm * 16 + lh * 8;
  float* __restrict__ op = Out + (size_t)mbase * NOUT + tg * (NT * 16) + lm;
  if (mbase + 8 <= M) {                                  // fast path: no guards
#pragma unroll
    for (int t = 0; t < NT; ++t)
#pragma unroll
      for (int r = 0; r < 8; ++r)
        op[r * NOUT + t * 16] = acc[t][r];
  } else {
#pragma unroll
    for (int t = 0; t < NT; ++t)
#pragma unroll
      for (int r = 0; r < 8; ++r)
        if (mbase + r < M) op[r * NOUT + t * 16] = acc[t][r];
  }
}

// ---------- attention logits: al_s[n,h] = sum_c h[n,h,c]*a_src[h,c] ----------
__global__ void k_attn_logits(const float* __restrict__ hf,
                              const float* __restrict__ asrc,
                              const float* __restrict__ adst,
                              float* __restrict__ als, float* __restrict__ ald,
                              int Nn, int H, int C) {
  int i = blockIdx.x * blockDim.x + threadIdx.x;
  if (i >= Nn * H) return;
  int n = i / H, h = i - n * H;
  const float* __restrict__ hp = hf + (size_t)n * H * C + (size_t)h * C;
  float s = 0.0f, d = 0.0f;
  for (int c = 0; c < C; ++c) {
    float v = hp[c];
    s += v * asrc[h * C + c];
    d += v * adst[h * C + c];
  }
  als[i] = s;
  ald[i] = d;
}

// ---------- edge pass 1: segment max of leaky-relu logits over dst ----------
__global__ void k_edge_max(const int* __restrict__ ei, int E, int ET,
                           const float* __restrict__ als, const float* __restrict__ ald,
                           u32* __restrict__ menc, int H) {
  int i = blockIdx.x * blockDim.x + threadIdx.x;
  if (i >= ET * H) return;
  int e = i / H, h = i - e * H;
  int s, d; edge_sd(ei, E, e, s, d);
  float v = als[s * H + h] + ald[d * H + h];
  v = v > 0.0f ? v : GAT_NEG_SLOPE * v;
  atomicMax(&menc[d * H + h], fenc(v));
}

// ---------- edge pass 2: denom[dst,h] += exp(e - m[dst,h]) ----------
__global__ void k_edge_denom(const int* __restrict__ ei, int E, int ET,
                             const float* __restrict__ als, const float* __restrict__ ald,
                             const u32* __restrict__ menc, float* __restrict__ denom, int H) {
  int i = blockIdx.x * blockDim.x + threadIdx.x;
  if (i >= ET * H) return;
  int e = i / H, h = i - e * H;
  int s, d; edge_sd(ei, E, e, s, d);
  float v = als[s * H + h] + ald[d * H + h];
  v = v > 0.0f ? v : GAT_NEG_SLOPE * v;
  atomicAdd(&denom[d * H + h], expf(v - fdec(menc[d * H + h])));
}

// ---------- edge pass 3: accum[dst] += alpha * h[src] ----------
// one block per edge; threads = H*C (<=256); per-head alpha staged in LDS.
// accum + feature tables both fit in MI455X's 192MB L2 -> atomics hit L2.
__global__ void k_edge_aggregate(const int* __restrict__ ei, int E, int ET,
                                 const float* __restrict__ als, const float* __restrict__ ald,
                                 const u32* __restrict__ menc, const float* __restrict__ denom,
                                 const float* __restrict__ hf, float* __restrict__ accum,
                                 int H, int C) {
  __shared__ float salpha[8];
  int e = blockIdx.x;
  if (e >= ET) return;
  int s, d; edge_sd(ei, E, e, s, d);
  int j = threadIdx.x;                  // 0 .. H*C-1
  if (j < H) {
    float v = als[s * H + j] + ald[d * H + j];
    v = v > 0.0f ? v : GAT_NEG_SLOPE * v;
    float ex = expf(v - fdec(menc[d * H + j]));
    salpha[j] = ex / denom[d * H + j];
  }
  __syncthreads();
  int h = j / C;
  atomicAdd(&accum[(size_t)d * H * C + j], hf[(size_t)s * H * C + j] * salpha[h]);
}

// ---------- finalize: bias (+ELU) then emit bf16 features for next GEMM ----------
__global__ void k_finalize_elu_bf16(const float* __restrict__ accum,
                                    const float* __restrict__ bias,
                                    u16* __restrict__ xbf, int total, int HC) {
  int i = blockIdx.x * blockDim.x + threadIdx.x;
  if (i >= total) return;
  float a = accum[i] + bias[i % HC];
  a = a > 0.0f ? a : (expf(a) - 1.0f);   // ELU
  xbf[i] = f2bf(a);
}
__global__ void k_finalize_out(const float* __restrict__ accum,
                               const float* __restrict__ bias,
                               float* __restrict__ out, int total, int HC) {
  int i = blockIdx.x * blockDim.x + threadIdx.x;
  if (i >= total) return;
  out[i] = accum[i] + bias[i % HC];      // H==1 -> mean over heads is identity
}

extern "C" void kernel_launch(void* const* d_in, const int* in_sizes, int n_in,
                              void* d_out, int out_size, void* d_ws, size_t ws_size,
                              hipStream_t stream) {
  const float* x   = (const float*)d_in[0];
  const int*   ei  = (const int*)d_in[1];
  const float* W1  = (const float*)d_in[2];
  const float* as1 = (const float*)d_in[3];
  const float* ad1 = (const float*)d_in[4];
  const float* b1  = (const float*)d_in[5];
  const float* W2  = (const float*)d_in[6];
  const float* as2 = (const float*)d_in[7];
  const float* ad2 = (const float*)d_in[8];
  const float* b2  = (const float*)d_in[9];
  const float* W3  = (const float*)d_in[10];
  const float* as3 = (const float*)d_in[11];
  const float* ad3 = (const float*)d_in[12];
  const float* b3  = (const float*)d_in[13];

  const int IN_C = 256, H = 8, C = 32, OUTC = 64;
  const int Nn = in_sizes[0] / IN_C;        // 50000
  const int E  = in_sizes[1] / 2;           // 800000
  const int ET = E + Nn;                    // + self loops

  // workspace carve-out (256B aligned), ~135 MB total
  char* w = (char*)d_ws;
  auto carve = [&](size_t bytes) -> char* {
    char* p = w; w += (bytes + 255) & ~(size_t)255; return p;
  };
  u16*   xbf = (u16*)  carve((size_t)Nn * 256 * 2);
  u32*   w1p = (u32*)  carve((size_t)(256 / 16) * (256 / 32) * 32 * 8 * 4);
  u32*   w2p = (u32*)  carve((size_t)(256 / 16) * (256 / 32) * 32 * 8 * 4);
  u32*   w3p = (u32*)  carve((size_t)(64 / 16) * (256 / 32) * 32 * 8 * 4);
  float* hfe = (float*)carve((size_t)Nn * 256 * 4);
  float* acc = (float*)carve((size_t)Nn * 256 * 4);
  float* als = (float*)carve((size_t)Nn * H * 4);
  float* ald = (float*)carve((size_t)Nn * H * 4);
  u32*  menc = (u32*)  carve((size_t)Nn * H * 4);
  float* den = (float*)carve((size_t)Nn * H * 4);

  auto cdiv = [](int a, int b) { return (a + b - 1) / b; };

  // convert features; pack weights into B-fragment layout (bf16)
  k_f32_to_bf16<<<cdiv(Nn * 256, 256), 256, 0, stream>>>(x, xbf, Nn * 256);
  k_pack_w<256, 256><<<cdiv(16 * 8 * 256, 256), 256, 0, stream>>>(W1, w1p);
  k_pack_w<256, 256><<<cdiv(16 * 8 * 256, 256), 256, 0, stream>>>(W2, w2p);
  k_pack_w<256, 64><<<cdiv(4 * 8 * 256, 256), 256, 0, stream>>>(W3, w3p);

  // edge/softmax/aggregate part (shared by all layers)
  auto edge_part = [&](int NOUT, int Hh, int Cc, const float* asr, const float* ads) {
    k_attn_logits<<<cdiv(Nn * Hh, 256), 256, 0, stream>>>(hfe, asr, ads, als, ald,
                                                          Nn, Hh, Cc);
    k_zero_f32<<<cdiv(Nn * NOUT, 256), 256, 0, stream>>>(acc, Nn * NOUT);
    k_zero_f32<<<cdiv(Nn * Hh, 256), 256, 0, stream>>>(den, Nn * Hh);
    k_init_enc<<<cdiv(Nn * Hh, 256), 256, 0, stream>>>(menc, Nn * Hh);
    k_edge_max<<<cdiv(ET * Hh, 256), 256, 0, stream>>>(ei, E, ET, als, ald, menc, Hh);
    k_edge_denom<<<cdiv(ET * Hh, 256), 256, 0, stream>>>(ei, E, ET, als, ald, menc,
                                                         den, Hh);
    k_edge_aggregate<<<ET, Hh * Cc, 0, stream>>>(ei, E, ET, als, ald, menc, den,
                                                 hfe, acc, Hh, Cc);
  };

  // GEMM wave counts: 16x64 strip per wave -> groupsN = NOUT/64 (min 1)
  const int waves256 = cdiv(Nn, 16) * (256 / 64);   // NT=4, tilesN=16
  const int waves64  = cdiv(Nn, 16) * 1;            // NT=4, tilesN=4

  // layer 1: [N,256] -> [N,8,32] concat -> ELU -> bf16
  k_gemm_bf16<256, 256><<<cdiv(waves256, 8), 256, 0, stream>>>(xbf, w1p, hfe, Nn);
  edge_part(256, H, C, as1, ad1);
  k_finalize_elu_bf16<<<cdiv(Nn * 256, 256), 256, 0, stream>>>(acc, b1, xbf,
                                                               Nn * 256, 256);
  // layer 2
  k_gemm_bf16<256, 256><<<cdiv(waves256, 8), 256, 0, stream>>>(xbf, w2p, hfe, Nn);
  edge_part(256, H, C, as2, ad2);
  k_finalize_elu_bf16<<<cdiv(Nn * 256, 256), 256, 0, stream>>>(acc, b2, xbf,
                                                               Nn * 256, 256);
  // layer 3: single head, C=64, no ELU, mean over 1 head == identity
  k_gemm_bf16<256, 64><<<cdiv(waves64, 8), 256, 0, stream>>>(xbf, w3p, hfe, Nn);
  edge_part(64, 1, OUTC, as3, ad3);
  k_finalize_out<<<cdiv(Nn * 64, 256), 256, 0, stream>>>(acc, b3, (float*)d_out,
                                                         Nn * 64, 64);
}
